// Qwen2_5VLVisionWrapper_76098230550797
// MI455X (gfx1250) — compile-verified, matrix operand order checked by reading, not run on hardware
//
#include <hip/hip_runtime.h>
#include <math.h>

// ---------------------------------------------------------------------------
// CDNA5 WMMA types (wave32, v_wmma_f32_16x16x32_bf16)
// ---------------------------------------------------------------------------
typedef __attribute__((ext_vector_type(16))) __bf16 v16bf;
typedef __attribute__((ext_vector_type(8)))  float  v8f;
typedef __attribute__((ext_vector_type(4)))  int    v4i;

#define S_TOK   1024
#define D_MODEL 1280
#define N_HEADS 16
#define HEAD_DIM 80
#define HD_PAD  96
#define D_FF    3456
#define DEPTH   8
#define OUT_DIM 3584

enum { EP_BIAS = 1, EP_RESID = 2, EP_SILU = 4, EP_GELU = 8 };

// float -> bf16 round-to-nearest-even
__device__ __forceinline__ __bf16 f2bf(float x) {
  union { float f; unsigned u; } a; a.f = x;
  unsigned r = a.u + 0x7FFFu + ((a.u >> 16) & 1u);
  unsigned short h = (unsigned short)(r >> 16);
  __bf16 o; __builtin_memcpy(&o, &h, 2); return o;
}

__device__ __forceinline__ v8f wmma_bf16(v16bf a, v16bf b, v8f c) {
  return __builtin_amdgcn_wmma_f32_16x16x32_bf16(false, a, false, b,
                                                 (short)0, c, false, false);
}

// ---------------------------------------------------------------------------
// CDNA5 async global->LDS copy (ASYNCcnt path).
// Confirmed signature: (global int4* src, local int4* dst, imm off, imm cpol)
// ---------------------------------------------------------------------------
__device__ __forceinline__ void cp_async_b128(void* lds_dst, const void* g_src) {
#if __has_builtin(__builtin_amdgcn_global_load_async_to_lds_b128)
  void* g = const_cast<void*>(g_src);
  __builtin_amdgcn_global_load_async_to_lds_b128(
      (__attribute__((address_space(1))) v4i*)g,
      (__attribute__((address_space(3))) v4i*)lds_dst, 0, 0);
#else
  unsigned lds_off = (unsigned)(unsigned long long)lds_dst;
  asm volatile("global_load_async_to_lds_b128 %0, %1, off"
               :: "v"(lds_off), "v"(g_src) : "memory");
#endif
}

__device__ __forceinline__ void wait_async_all() {
#if __has_builtin(__builtin_amdgcn_s_wait_asynccnt)
  __builtin_amdgcn_s_wait_asynccnt(0);
#else
  asm volatile("s_wait_asynccnt 0x0" ::: "memory");
#endif
}

// 16-bit operand fragment from row-major bf16 in global memory.
// Element e -> k = (e/8)*16 + (lane>=16?8:0) + e%8 (CDNA5 16-bit A/B layout).
__device__ __forceinline__ v16bf frag_ld(const __bf16* __restrict__ base,
                                         unsigned off, int lane) {
  const __bf16* p = base + off + (unsigned)(((lane >> 4) & 1) * 8);
  union { uint4 u[2]; v16bf v; } f;
  f.u[0] = *(const uint4*)(p);
  f.u[1] = *(const uint4*)(p + 16);
  return f.v;
}

// Fragment read from an LDS-staged packed panel: lane's 32B are contiguous.
__device__ __forceinline__ v16bf frag_lds(const void* buf, int t, int lane) {
  const uint4* p = (const uint4*)((const char*)buf + t * 1024 + lane * 32);
  union { uint4 u[2]; v16bf v; } f;
  f.u[0] = p[0];
  f.u[1] = p[1];
  return f.v;
}

// ---------------------------------------------------------------------------
// Weight repack: fp32 W[K x N] -> bf16 fragment-packed panels.
// Tile (nt, kt) of 32(K) x 16(N); storage ((nt*KT + kt)*32 + lane)*16 + e,
// so one 32x64 k-panel of a block is four contiguous 1KB chunks.
// ---------------------------------------------------------------------------
__global__ void repack_w(const float* __restrict__ W, __bf16* __restrict__ P,
                         int K, int Kvalid, int N) {
  unsigned idx = blockIdx.x * 256u + threadIdx.x;
  unsigned KT = (unsigned)K >> 5;
  unsigned total = ((unsigned)N >> 4) * KT * 32u;
  if (idx >= total) return;
  int lane = idx & 31;
  unsigned t = idx >> 5;
  unsigned kt = t % KT;
  unsigned nt = t / KT;
  unsigned col = nt * 16u + (lane & 15);
  int kb = (int)(kt * 32u) + ((lane >> 4) & 1) * 8;
  union { uint4 u[2]; __bf16 e[16]; } f;
#pragma unroll
  for (int g = 0; g < 2; ++g)
#pragma unroll
    for (int i = 0; i < 8; ++i) {
      int k = kb + g * 16 + i;
      f.e[g * 8 + i] = (k < Kvalid) ? f2bf(W[(unsigned)k * (unsigned)N + col])
                                    : f2bf(0.f);
    }
  uint4* dst = (uint4*)(P + (size_t)idx * 16);
  dst[0] = f.u[0];
  dst[1] = f.u[1];
}

// ---------------------------------------------------------------------------
// GEMM: D[S x N] = epilogue( A_bf16[S x K] * Bpacked )
// 256 threads = 8 waves; wave -> 16 rows x 64 cols (4 WMMA accumulators).
// B panel (4KB per k-step) staged once per block into LDS via async copy,
// double-buffered; waves consume it with ds_load_b128.
// grid = (N/64, S/128). K multiple of 32, N multiple of 64.
// ---------------------------------------------------------------------------
template <int FLAGS>
__global__ void gemm_bf16_wmma(const __bf16* __restrict__ A,
                               const __bf16* __restrict__ Bp,
                               const float* __restrict__ bias,
                               const float* __restrict__ resid,
                               float* __restrict__ Df,
                               __bf16* __restrict__ Dbf,
                               int K, int N) {
  const int lane = threadIdx.x & 31;
  const int wave = threadIdx.x >> 5;
  const int mBase = blockIdx.y * 128 + wave * 16;
  const int nBase = blockIdx.x * 64;
  const unsigned KT = (unsigned)K >> 5;
  const unsigned ntStride = KT * 512u;          // elements per n-tile panel

  __shared__ __align__(16) unsigned char smem[8192];
  char* sm = (char*)smem;                       // runtime addrspace cast only

  const unsigned chunk = threadIdx.x >> 6;      // n-tile 0..3 for staging
  const unsigned unit  = threadIdx.x & 63u;     // 16B unit within 1KB chunk
  const unsigned stageOff = chunk * 1024u + unit * 16u;
  const unsigned bPanel0 = (unsigned)(nBase >> 4) * KT * 512u;
  const __bf16* gStage = Bp + bPanel0 + chunk * ntStride + unit * 8u;

  // Prologue: stage panel kt=0 into buffer 0
  cp_async_b128(sm + stageOff, gStage);
  wait_async_all();
  __syncthreads();

  unsigned aOff = (unsigned)(mBase + (lane & 15)) * (unsigned)K;

  v8f acc0 = {0,0,0,0,0,0,0,0}, acc1 = {0,0,0,0,0,0,0,0};
  v8f acc2 = {0,0,0,0,0,0,0,0}, acc3 = {0,0,0,0,0,0,0,0};

  for (unsigned kt = 0; kt < KT; ++kt) {
    char* cur = sm + ((kt & 1u) * 4096u);
    if (kt + 1u < KT)   // kick off next panel while we compute on this one
      cp_async_b128(sm + (((kt + 1u) & 1u) * 4096u) + stageOff,
                    gStage + (kt + 1u) * 512u);
    v16bf a = frag_ld(A, aOff, lane);
    aOff += 32u;
    acc0 = wmma_bf16(a, frag_lds(cur, 0, lane), acc0);
    acc1 = wmma_bf16(a, frag_lds(cur, 1, lane), acc1);
    acc2 = wmma_bf16(a, frag_lds(cur, 2, lane), acc2);
    acc3 = wmma_bf16(a, frag_lds(cur, 3, lane), acc3);
    if (kt + 1u < KT) {
      wait_async_all();   // own wave's async writes done
      __syncthreads();    // all waves' writes done; all reads of cur done
    }
  }

  v8f accs[4] = {acc0, acc1, acc2, acc3};
  unsigned row0 = (unsigned)(mBase + ((lane >> 4) & 1) * 8);
#pragma unroll
  for (int t = 0; t < 4; ++t) {
    unsigned c = (unsigned)(nBase + t * 16 + (lane & 15));
    float bv = (FLAGS & EP_BIAS) ? bias[c] : 0.f;
#pragma unroll
    for (int r = 0; r < 8; ++r) {
      float x = accs[t][r] + bv;
      unsigned idx = (row0 + (unsigned)r) * (unsigned)N + c;
      if (FLAGS & EP_RESID) x += resid[idx];
      if (FLAGS & EP_SILU)  x = x / (1.f + __expf(-x));
      if (FLAGS & EP_GELU)  x = 0.5f * x * (1.f + erff(x * 0.70710678118654752f));
      if (Df)  Df[idx]  = x;
      if (Dbf) Dbf[idx] = f2bf(x);
    }
  }
}

// ---------------------------------------------------------------------------
// Attention scores: one wave -> 16 queries x 64 keys for one head.
// ---------------------------------------------------------------------------
__global__ void attn_scores(const __bf16* __restrict__ Qb,
                            const __bf16* __restrict__ Kb,
                            float* __restrict__ scores,
                            int Lk, int windowed) {
  int lane = threadIdx.x & 31;
  int wave = threadIdx.x >> 5;
  int tile = blockIdx.x * 8 + wave;
  int kgPer = Lk >> 6;
  int head = tile / (64 * kgPer);
  int rem  = tile % (64 * kgPer);
  int qt = rem / kgPer;
  int kg = rem % kgPer;
  int qBase = qt * 16;
  int keyBase = windowed ? (qBase & ~63) : kg * 64;

  unsigned qOff0 = (unsigned)head * (S_TOK * HD_PAD)
                 + (unsigned)(qBase + (lane & 15)) * HD_PAD;
  unsigned kOff0 = (unsigned)head * (S_TOK * HD_PAD)
                 + (unsigned)(keyBase + (lane & 15)) * HD_PAD;

  v8f acc0 = {0,0,0,0,0,0,0,0}, acc1 = {0,0,0,0,0,0,0,0};
  v8f acc2 = {0,0,0,0,0,0,0,0}, acc3 = {0,0,0,0,0,0,0,0};
#pragma unroll
  for (int dt = 0; dt < HD_PAD; dt += 32) {
    v16bf a = frag_ld(Qb, qOff0 + (unsigned)dt, lane);
    acc0 = wmma_bf16(a, frag_ld(Kb, kOff0 + (unsigned)dt, lane), acc0);
    acc1 = wmma_bf16(a, frag_ld(Kb, kOff0 + 16u * HD_PAD + (unsigned)dt, lane), acc1);
    acc2 = wmma_bf16(a, frag_ld(Kb, kOff0 + 32u * HD_PAD + (unsigned)dt, lane), acc2);
    acc3 = wmma_bf16(a, frag_ld(Kb, kOff0 + 48u * HD_PAD + (unsigned)dt, lane), acc3);
  }

  v8f accs[4] = {acc0, acc1, acc2, acc3};
  unsigned base = (unsigned)head * (unsigned)(S_TOK * Lk);
  unsigned row0 = (unsigned)(qBase + ((lane >> 4) & 1) * 8);
  unsigned colb = (unsigned)((windowed ? 0 : keyBase) + (lane & 15));
#pragma unroll
  for (int t = 0; t < 4; ++t)
#pragma unroll
    for (int r = 0; r < 8; ++r)
      scores[base + (row0 + (unsigned)r) * (unsigned)Lk + colb + (unsigned)(t * 16)]
          = accs[t][r];
}

// scale + softmax over Lk keys -> bf16 probs. One 128-thr block per row.
__global__ void softmax_rows(const float* __restrict__ scores,
                             __bf16* __restrict__ probs, int Lk, float scale) {
  unsigned row = blockIdx.x;
  const float* s = scores + (size_t)row * Lk;
  __bf16* p = probs + (size_t)row * Lk;
  __shared__ float red[128];

  float m = -1e30f;
  for (int c = threadIdx.x; c < Lk; c += 128) m = fmaxf(m, s[c] * scale);
  red[threadIdx.x] = m; __syncthreads();
  for (int off = 64; off; off >>= 1) {
    if (threadIdx.x < (unsigned)off)
      red[threadIdx.x] = fmaxf(red[threadIdx.x], red[threadIdx.x + off]);
    __syncthreads();
  }
  m = red[0]; __syncthreads();

  float sum = 0.f;
  for (int c = threadIdx.x; c < Lk; c += 128) sum += __expf(s[c] * scale - m);
  red[threadIdx.x] = sum; __syncthreads();
  for (int off = 64; off; off >>= 1) {
    if (threadIdx.x < (unsigned)off) red[threadIdx.x] += red[threadIdx.x + off];
    __syncthreads();
  }
  float inv = 1.f / red[0];
  for (int c = threadIdx.x; c < Lk; c += 128)
    p[c] = f2bf(__expf(s[c] * scale - m) * inv);
}

// O = probs @ V : one wave -> 16 queries x 80 dims for one head.
// V staged transposed [head][dim][token] so K runs contiguous.
__global__ void attn_o(const __bf16* __restrict__ probs,
                       const __bf16* __restrict__ Vt,
                       __bf16* __restrict__ obf, int Lk) {
  int lane = threadIdx.x & 31;
  int wave = threadIdx.x >> 5;
  int tile = blockIdx.x * 8 + wave;      // 16 heads * 64 qtiles
  int head = tile / 64, qt = tile % 64;
  int qBase = qt * 16;
  int keyBase = (Lk == 64) ? (qBase & ~63) : 0;

  unsigned pOff0 = ((unsigned)head * S_TOK + (unsigned)(qBase + (lane & 15)))
                 * (unsigned)Lk;
  unsigned vOff0 = (unsigned)head * (HEAD_DIM * S_TOK)
                 + (unsigned)(lane & 15) * S_TOK + (unsigned)keyBase;

  v8f acc0 = {0,0,0,0,0,0,0,0}, acc1 = {0,0,0,0,0,0,0,0};
  v8f acc2 = {0,0,0,0,0,0,0,0}, acc3 = {0,0,0,0,0,0,0,0};
  v8f acc4 = {0,0,0,0,0,0,0,0};
  for (int kt = 0; kt < Lk; kt += 32) {
    v16bf a = frag_ld(probs, pOff0 + (unsigned)kt, lane);
    unsigned vo = vOff0 + (unsigned)kt;
    acc0 = wmma_bf16(a, frag_ld(Vt, vo, lane), acc0);
    acc1 = wmma_bf16(a, frag_ld(Vt, vo + 16u * S_TOK, lane), acc1);
    acc2 = wmma_bf16(a, frag_ld(Vt, vo + 32u * S_TOK, lane), acc2);
    acc3 = wmma_bf16(a, frag_ld(Vt, vo + 48u * S_TOK, lane), acc3);
    acc4 = wmma_bf16(a, frag_ld(Vt, vo + 64u * S_TOK, lane), acc4);
  }
  v8f accs[5] = {acc0, acc1, acc2, acc3, acc4};
  unsigned row0 = (unsigned)(qBase + ((lane >> 4) & 1) * 8);
  unsigned col0 = (unsigned)(head * HEAD_DIM + (lane & 15));
#pragma unroll
  for (int t = 0; t < 5; ++t)
#pragma unroll
    for (int r = 0; r < 8; ++r)
      obf[(row0 + (unsigned)r) * D_MODEL + col0 + (unsigned)(t * 16)]
          = f2bf(accs[t][r]);
}

// ---------------------------------------------------------------------------
// Elementwise / staging kernels
// ---------------------------------------------------------------------------
__global__ void rmsnorm_k(const float* __restrict__ x, const float* __restrict__ w,
                          __bf16* __restrict__ out, int D) {
  unsigned row = blockIdx.x;
  const float* xr = x + (size_t)row * D;
  __shared__ float red[256];
  float s = 0.f;
  for (int c = threadIdx.x; c < D; c += 256) { float v = xr[c]; s += v * v; }
  red[threadIdx.x] = s; __syncthreads();
  for (int off = 128; off; off >>= 1) {
    if (threadIdx.x < (unsigned)off) red[threadIdx.x] += red[threadIdx.x + off];
    __syncthreads();
  }
  float inv = rsqrtf(red[0] / (float)D + 1e-6f);
  for (int c = threadIdx.x; c < D; c += 256)
    out[row * (unsigned)D + c] = f2bf(xr[c] * inv * w[c]);
}

// images (1,3,448,448) -> bf16 patch vectors (1024 x 1184), cols>=1176 zero.
__global__ void preprocess_k(const float* __restrict__ images, __bf16* __restrict__ pv) {
  unsigned idx = blockIdx.x * 256u + threadIdx.x;
  if (idx >= 1024u * 1184u) return;
  unsigned row = idx / 1184u, col = idx % 1184u;
  float val = 0.f;
  if (col < 1176u) {
    unsigned ct = col / 196u, rr = col % 196u, ph = rr / 14u, pw = rr % 14u;
    unsigned c = ct >> 1;
    unsigned gb = row >> 2, mm = row & 3u, mh = mm >> 1, mw = mm & 1u;
    unsigned ghb = gb / 16u, gwb = gb % 16u;
    unsigned h = (ghb * 2u + mh) * 14u + ph;
    unsigned w = (gwb * 2u + mw) * 14u + pw;
    val = images[(c * 448u + h) * 448u + w];
  }
  pv[idx] = f2bf(val);
}

__global__ void permute_rows(const float* __restrict__ src, const int* __restrict__ win,
                             float* __restrict__ dst, int D) {
  unsigned t = blockIdx.x;
  unsigned srow = (unsigned)win[t >> 2] * 4u + (t & 3u);
  for (int c = threadIdx.x; c < D; c += 256)
    dst[t * (unsigned)D + c] = src[srow * (unsigned)D + c];
}

__global__ void rope_tables(const float* __restrict__ rpe, const int* __restrict__ win,
                            float* __restrict__ cosb, float* __restrict__ sinb) {
  unsigned idx = blockIdx.x * 256u + threadIdx.x;
  if (idx >= S_TOK * HEAD_DIM) return;
  unsigned t = idx / HEAD_DIM, d = idx % HEAD_DIM;
  unsigned srow = (unsigned)win[t >> 2] * 4u + (t & 3u);
  float a = rpe[srow * 40u + (d % 40u)];
  cosb[idx] = cosf(a);
  sinb[idx] = sinf(a);
}

// qkv fp32 (S x 3840) -> rotated Q/K bf16 [head][tok][96], V^T bf16 [head][dim][tok]
__global__ void rope_stage(const float* __restrict__ qkv,
                           const float* __restrict__ cosb, const float* __restrict__ sinb,
                           __bf16* __restrict__ Qb, __bf16* __restrict__ Kb,
                           __bf16* __restrict__ Vt) {
  unsigned idx = blockIdx.x * 256u + threadIdx.x;
  if (idx >= S_TOK * N_HEADS * HD_PAD) return;
  unsigned d = idx % HD_PAD;
  unsigned th = idx / HD_PAD;
  unsigned head = th % N_HEADS;
  unsigned t = th / N_HEADS;
  unsigned qi = (head * S_TOK + t) * HD_PAD + d;
  if (d >= HEAD_DIM) { Qb[qi] = f2bf(0.f); Kb[qi] = f2bf(0.f); return; }
  const float* row = qkv + t * (3u * D_MODEL);
  float c = cosb[t * HEAD_DIM + d], s = sinb[t * HEAD_DIM + d];
  unsigned qo = head * HEAD_DIM + d;
  float q = row[qo];
  float k = row[D_MODEL + qo];
  float v = row[2u * D_MODEL + qo];
  float qh = (d < 40u) ? -row[qo + 40u] : row[qo - 40u];
  float kh = (d < 40u) ? -row[D_MODEL + qo + 40u] : row[D_MODEL + qo - 40u];
  Qb[qi] = f2bf(q * c + qh * s);
  Kb[qi] = f2bf(k * c + kh * s);
  Vt[(head * HEAD_DIM + d) * S_TOK + t] = f2bf(v);
}

__global__ void mul_bf16(const float* __restrict__ g, const float* __restrict__ u,
                         __bf16* __restrict__ out, int n) {
  unsigned i = blockIdx.x * 256u + threadIdx.x;
  if (i < (unsigned)n) out[i] = f2bf(g[i] * u[i]);
}

__global__ void gather_out(const float* __restrict__ m, const int* __restrict__ rev,
                           float* __restrict__ out) {
  unsigned r = blockIdx.x;
  unsigned src = (unsigned)rev[r];
  for (int c = threadIdx.x; c < OUT_DIM; c += 256)
    out[r * (unsigned)OUT_DIM + c] = m[src * (unsigned)OUT_DIM + c];
}

// ---------------------------------------------------------------------------
// Host orchestration
// ---------------------------------------------------------------------------
static inline void repack(const float* W, __bf16* P, int K, int Kvalid, int N,
                          hipStream_t s) {
  unsigned total = (unsigned)(N >> 4) * (unsigned)(K >> 5) * 32u;
  repack_w<<<(total + 255u) / 256u, 256, 0, s>>>(W, P, K, Kvalid, N);
}

extern "C" void kernel_launch(void* const* d_in, const int* in_sizes, int n_in,
                              void* d_out, int out_size, void* d_ws, size_t ws_size,
                              hipStream_t stream) {
  (void)in_sizes; (void)n_in; (void)out_size; (void)ws_size;
  const float* images  = (const float*)d_in[0];
  const int*   win     = (const int*)  d_in[1];
  const int*   rev     = (const int*)  d_in[2];
  const float* rpe     = (const float*)d_in[3];
  const float* patch_w = (const float*)d_in[6];
  const float* ln1_w   = (const float*)d_in[7];
  const float* ln2_w   = (const float*)d_in[8];
  const float* qkv_w   = (const float*)d_in[9];
  const float* qkv_b   = (const float*)d_in[10];
  const float* proj_w  = (const float*)d_in[11];
  const float* proj_b  = (const float*)d_in[12];
  const float* gate_w  = (const float*)d_in[13];
  const float* gate_b  = (const float*)d_in[14];
  const float* up_w    = (const float*)d_in[15];
  const float* up_b    = (const float*)d_in[16];
  const float* down_w  = (const float*)d_in[17];
  const float* down_b  = (const float*)d_in[18];
  const float* lnq_w   = (const float*)d_in[19];
  const float* m1_w    = (const float*)d_in[20];
  const float* m1_b    = (const float*)d_in[21];
  const float* m2_w    = (const float*)d_in[22];
  const float* m2_b    = (const float*)d_in[23];
  float* out = (float*)d_out;

  size_t off = 0;
  auto alloc = [&](size_t bytes) -> void* {
    void* p = (char*)d_ws + off;
    off = (off + bytes + 255) & ~(size_t)255;
    return p;
  };
  float*  cosb   = (float*) alloc((size_t)S_TOK * HEAD_DIM * 4);
  float*  sinb   = (float*) alloc((size_t)S_TOK * HEAD_DIM * 4);
  __bf16* pv     = (__bf16*)alloc((size_t)S_TOK * 1184 * 2);
  float*  etmp   = (float*) alloc((size_t)S_TOK * D_MODEL * 4);
  float*  h      = (float*) alloc((size_t)S_TOK * D_MODEL * 4);
  __bf16* anorm  = (__bf16*)alloc((size_t)S_TOK * D_MODEL * 2);
  float*  qkvbuf = (float*) alloc((size_t)S_TOK * 3 * D_MODEL * 4);
  __bf16* Qb     = (__bf16*)alloc((size_t)N_HEADS * S_TOK * HD_PAD * 2);
  __bf16* Kb     = (__bf16*)alloc((size_t)N_HEADS * S_TOK * HD_PAD * 2);
  __bf16* Vt     = (__bf16*)alloc((size_t)N_HEADS * HEAD_DIM * S_TOK * 2);
  float*  scores = (float*) alloc((size_t)N_HEADS * S_TOK * S_TOK * 4);
  __bf16* probs  = (__bf16*)alloc((size_t)N_HEADS * S_TOK * S_TOK * 2);
  __bf16* obf    = (__bf16*)alloc((size_t)S_TOK * D_MODEL * 2);
  float*  gout   = (float*) alloc((size_t)S_TOK * D_FF * 4);
  float*  uout   = (float*) alloc((size_t)S_TOK * D_FF * 4);
  __bf16* midbf  = (__bf16*)alloc((size_t)S_TOK * D_FF * 2);
  __bf16* m1bf   = (__bf16*)alloc((size_t)256 * 5120 * 2);
  float*  m2out  = (float*) alloc((size_t)256 * OUT_DIM * 4);
  __bf16* packw  = (__bf16*)alloc((size_t)5120 * 5120 * 2);   // largest panel

  const float scale = 0.11180339887498949f; // 1/sqrt(80)

  // Patch embed
  preprocess_k<<<(1024 * 1184 + 255) / 256, 256, 0, stream>>>(images, pv);
  repack(patch_w, packw, 1184, 1176, D_MODEL, stream);
  gemm_bf16_wmma<0><<<dim3(D_MODEL / 64, S_TOK / 128), 256, 0, stream>>>(
      pv, packw, nullptr, nullptr, etmp, nullptr, 1184, D_MODEL);
  permute_rows<<<S_TOK, 256, 0, stream>>>(etmp, win, h, D_MODEL);
  rope_tables<<<(S_TOK * HEAD_DIM + 255) / 256, 256, 0, stream>>>(rpe, win, cosb, sinb);

  for (int i = 0; i < DEPTH; ++i) {
    int full = (i == 3 || i == 7);
    int Lk = full ? S_TOK : 64;

    // ---- attention ----
    rmsnorm_k<<<S_TOK, 256, 0, stream>>>(h, ln1_w + (size_t)i * D_MODEL, anorm, D_MODEL);
    repack(qkv_w + (size_t)i * D_MODEL * 3 * D_MODEL, packw, D_MODEL, D_MODEL,
           3 * D_MODEL, stream);
    gemm_bf16_wmma<EP_BIAS><<<dim3(3 * D_MODEL / 64, S_TOK / 128), 256, 0, stream>>>(
        anorm, packw, qkv_b + (size_t)i * 3 * D_MODEL, nullptr, qkvbuf, nullptr,
        D_MODEL, 3 * D_MODEL);
    rope_stage<<<(S_TOK * N_HEADS * HD_PAD + 255) / 256, 256, 0, stream>>>(
        qkvbuf, cosb, sinb, Qb, Kb, Vt);

    int tiles = N_HEADS * 64 * (Lk / 64);
    attn_scores<<<tiles / 8, 256, 0, stream>>>(Qb, Kb, scores, Lk, !full);
    softmax_rows<<<N_HEADS * S_TOK, 128, 0, stream>>>(scores, probs, Lk, scale);
    attn_o<<<(N_HEADS * 64) / 8, 256, 0, stream>>>(probs, Vt, obf, Lk);

    repack(proj_w + (size_t)i * D_MODEL * D_MODEL, packw, D_MODEL, D_MODEL,
           D_MODEL, stream);
    gemm_bf16_wmma<EP_BIAS | EP_RESID>
        <<<dim3(D_MODEL / 64, S_TOK / 128), 256, 0, stream>>>(
        obf, packw, proj_b + (size_t)i * D_MODEL, h, h, nullptr, D_MODEL, D_MODEL);

    // ---- MLP ----
    rmsnorm_k<<<S_TOK, 256, 0, stream>>>(h, ln2_w + (size_t)i * D_MODEL, anorm, D_MODEL);
    repack(gate_w + (size_t)i * D_MODEL * D_FF, packw, D_MODEL, D_MODEL, D_FF, stream);
    gemm_bf16_wmma<EP_BIAS | EP_SILU>
        <<<dim3(D_FF / 64, S_TOK / 128), 256, 0, stream>>>(
        anorm, packw, gate_b + (size_t)i * D_FF, nullptr, gout, nullptr,
        D_MODEL, D_FF);
    repack(up_w + (size_t)i * D_MODEL * D_FF, packw, D_MODEL, D_MODEL, D_FF, stream);
    gemm_bf16_wmma<EP_BIAS><<<dim3(D_FF / 64, S_TOK / 128), 256, 0, stream>>>(
        anorm, packw, up_b + (size_t)i * D_FF, nullptr, uout, nullptr,
        D_MODEL, D_FF);
    mul_bf16<<<(S_TOK * D_FF + 255) / 256, 256, 0, stream>>>(gout, uout, midbf,
                                                             S_TOK * D_FF);
    repack(down_w + (size_t)i * D_FF * D_MODEL, packw, D_FF, D_FF, D_MODEL, stream);
    gemm_bf16_wmma<EP_BIAS | EP_RESID>
        <<<dim3(D_MODEL / 64, S_TOK / 128), 256, 0, stream>>>(
        midbf, packw, down_b + (size_t)i * D_MODEL, h, h, nullptr, D_FF, D_MODEL);
  }

  // Merger: rms_norm over D_MODEL, then view (256 x 5120) contiguously
  rmsnorm_k<<<S_TOK, 256, 0, stream>>>(h, lnq_w, anorm, D_MODEL);
  repack(m1_w, packw, 5120, 5120, 5120, stream);
  gemm_bf16_wmma<EP_BIAS | EP_GELU><<<dim3(5120 / 64, 256 / 128), 256, 0, stream>>>(
      anorm, packw, m1_b, nullptr, nullptr, m1bf, 5120, 5120);
  repack(m2_w, packw, 5120, 5120, OUT_DIM, stream);
  gemm_bf16_wmma<EP_BIAS><<<dim3(OUT_DIM / 64, 256 / 128), 256, 0, stream>>>(
      m1bf, packw, m2_b, nullptr, m2out, nullptr, 5120, OUT_DIM);
  gather_out<<<256, 256, 0, stream>>>(m2out, rev, out);
}
